// MambaBlock_35802847379771
// MI455X (gfx1250) — matrix operand order repaired and checked by reading, not compiled, not run on hardware
//
#include <hip/hip_runtime.h>

// ---------------------------------------------------------------------------
// MambaBlock for MI455X (gfx1250, wave32, WMMA, async-to-LDS).
//   K1: inp  = x @ W_in^T + b_in                    (bf16 WMMA, f32 acc)
//   K2: gate = sigmoid(inp @ W_gate^T + b_gate)      \ fused, shared A frag
//       u    = inp @ B^T                             /
//   K3: recurrence, 16 persistent WGs, A resident in LDS, S staged per step
//   K4: out  = states @ W_out^T + b_out
// GEMMs double-buffer the 64x32 weight tile in LDS via
// global_load_async_to_lds_b128 (ASYNCcnt). B fragments of a k-step are all
// loaded into distinct registers before the WMMA batch (partial dscnt waits),
// and the recurrence uses two interleaved accumulators to break the WMMA
// RAW chain on the sequential critical path.
// ---------------------------------------------------------------------------

typedef __attribute__((ext_vector_type(16))) __bf16 v16bf;
typedef __attribute__((ext_vector_type(8)))  __bf16 v8bf;
typedef __attribute__((ext_vector_type(8)))  float  v8f;

#define DIM_   1024
#define SEQ_   4096
#define BATCH_ 8
#define NWG_REC 16
#define BT_STRIDE 40      // padded row stride (elems) for 64x32 LDS B tiles
#define AS_STRIDE 1032    // padded row stride (elems) for Asub / Slds rows

__device__ __forceinline__ v16bf ld_pair(const __bf16* p0, const __bf16* p1) {
  union { v16bf v; v8bf h[2]; } u;
  u.h[0] = *(const v8bf*)p0;      // 16B-aligned -> b128 (global or ds)
  u.h[1] = *(const v8bf*)p1;
  return u.v;
}

__device__ __forceinline__ v8f wmma_bf16(v16bf a, v16bf b, v8f c) {
  return __builtin_amdgcn_wmma_f32_16x16x32_bf16(false, a, false, b,
                                                 (short)0, c, false, false);
}

__device__ __forceinline__ v8f vzero8() {
  v8f z = {0.f,0.f,0.f,0.f,0.f,0.f,0.f,0.f};
  return z;
}

// Async 16B global->LDS copy (per-lane), tracked by ASYNCcnt.
// LDS byte address = low 32 bits of the generic shared pointer (aperture rule).
__device__ __forceinline__ void async_copy16(unsigned lds_off, const void* g) {
  asm volatile("global_load_async_to_lds_b128 %0, %1, off"
               :: "v"(lds_off), "v"((unsigned long long)(uintptr_t)g)
               : "memory");
}
__device__ __forceinline__ void wait_async0() {
  asm volatile("s_wait_asynccnt 0" ::: "memory");
}
__device__ __forceinline__ unsigned lds_addr(const void* p) {
  return (unsigned)(uintptr_t)p;
}

// ---------------- f32 -> bf16 convert --------------------------------------
__global__ void k_cvt_bf16(const float* __restrict__ in, void* outv, int n) {
  __bf16* out = (__bf16*)outv;
  int i = blockIdx.x * 256 + threadIdx.x;
  if (i < n) out[i] = (__bf16)in[i];
}

// ---------------- zero recurrence state + barrier counter ------------------
__global__ void k_init(void* Sv, unsigned* cnt) {
  __bf16* S = (__bf16*)Sv;
  int i = blockIdx.x * 256 + threadIdx.x;
  if (i < 2 * 16 * DIM_) S[i] = (__bf16)0.f;   // ping-pong 16x1024 buffers
  if (i == 0) *cnt = 0u;
}

// ---------------- GEMM: out = X @ W^T + bias  (bf16 or f32 store) ----------
// grid (T/128, D/64), 256 threads (8 waves); wave -> 16 rows x 64 cols.
// B tile (64x32 bf16) double-buffered in LDS via async copies.
template <bool STORE_BF16>
__global__ void k_gemm(const void* Xv, const void* Wv,
                       const float* __restrict__ bias, void* Ov) {
  __shared__ __bf16 Bt[2][64 * BT_STRIDE];
  const __bf16* __restrict__ X = (const __bf16*)Xv;
  const __bf16* __restrict__ W = (const __bf16*)Wv;
  const int wave = threadIdx.x >> 5, lane = threadIdx.x & 31;
  const int lrow = lane & 15, khalf = lane >> 4;
  const int m0 = blockIdx.x * 128 + wave * 16;
  const int n0 = blockIdx.y * 64;
  const __bf16* arow = X + (size_t)(m0 + lrow) * DIM_ + khalf * 8;

  // staging: thread -> (row, 8-elem chunk) of the 64x32 tile
  const int trow = threadIdx.x >> 2, tch = threadIdx.x & 3;
  const __bf16* gsrc = W + (size_t)(n0 + trow) * DIM_ + tch * 8;
  const unsigned myoff = (unsigned)((trow * BT_STRIDE + tch * 8) * 2);

  async_copy16(lds_addr(&Bt[0][0]) + myoff, gsrc);       // tile 0
  wait_async0();
  __syncthreads();

  v8f acc[4] = {vzero8(), vzero8(), vzero8(), vzero8()};
  int cur = 0;
  for (int k0 = 0; k0 < DIM_; k0 += 32) {
    if (k0 + 32 < DIM_)                                  // prefetch next tile
      async_copy16(lds_addr(&Bt[cur ^ 1][0]) + myoff, gsrc + k0 + 32);
    v16bf a = ld_pair(arow + k0, arow + k0 + 16);        // A frag: global
    const __bf16* bt = &Bt[cur][0];
    v16bf b[4];
#pragma unroll
    for (int j = 0; j < 4; ++j) {                        // all loads first
      const __bf16* bp = bt + (j * 16 + lrow) * BT_STRIDE + khalf * 8;
      b[j] = ld_pair(bp, bp + 16);
    }
#pragma unroll
    for (int j = 0; j < 4; ++j)                          // then WMMA batch
      acc[j] = wmma_bf16(a, b[j], acc[j]);
    wait_async0();                                       // own prefetch done
    __syncthreads();                                     // tile ready, reads done
    cur ^= 1;
  }

  const int rbase = khalf * 8;
#pragma unroll
  for (int j = 0; j < 4; ++j) {
    const int col = n0 + j * 16 + lrow;
    const float bv = bias[col];
#pragma unroll
    for (int r = 0; r < 8; ++r) {
      size_t idx = (size_t)(m0 + rbase + r) * DIM_ + col;
      if (STORE_BF16) ((__bf16*)Ov)[idx] = (__bf16)(acc[j][r] + bv);
      else            ((float*)Ov)[idx]  = acc[j][r] + bv;
    }
  }
}

// ---------------- fused GEMM: gate = sigmoid(X@Wg^T+bg), u = X@Bm^T --------
__global__ void k_gemm_dual(const void* Xv, const void* Wgv, const void* Bmv,
                            const float* __restrict__ bgate,
                            float* __restrict__ gate, float* __restrict__ u) {
  __shared__ __bf16 Btg[2][64 * BT_STRIDE];
  __shared__ __bf16 Btu[2][64 * BT_STRIDE];
  const __bf16* __restrict__ X  = (const __bf16*)Xv;
  const __bf16* __restrict__ Wg = (const __bf16*)Wgv;
  const __bf16* __restrict__ Bm = (const __bf16*)Bmv;
  const int wave = threadIdx.x >> 5, lane = threadIdx.x & 31;
  const int lrow = lane & 15, khalf = lane >> 4;
  const int m0 = blockIdx.x * 128 + wave * 16;
  const int n0 = blockIdx.y * 64;
  const __bf16* arow = X + (size_t)(m0 + lrow) * DIM_ + khalf * 8;

  const int trow = threadIdx.x >> 2, tch = threadIdx.x & 3;
  const size_t goff = (size_t)(n0 + trow) * DIM_ + tch * 8;
  const unsigned myoff = (unsigned)((trow * BT_STRIDE + tch * 8) * 2);

  async_copy16(lds_addr(&Btg[0][0]) + myoff, Wg + goff);
  async_copy16(lds_addr(&Btu[0][0]) + myoff, Bm + goff);
  wait_async0();
  __syncthreads();

  v8f accg[4] = {vzero8(), vzero8(), vzero8(), vzero8()};
  v8f accu[4] = {vzero8(), vzero8(), vzero8(), vzero8()};
  int cur = 0;
  for (int k0 = 0; k0 < DIM_; k0 += 32) {
    if (k0 + 32 < DIM_) {
      async_copy16(lds_addr(&Btg[cur ^ 1][0]) + myoff, Wg + goff + k0 + 32);
      async_copy16(lds_addr(&Btu[cur ^ 1][0]) + myoff, Bm + goff + k0 + 32);
    }
    v16bf a = ld_pair(arow + k0, arow + k0 + 16);        // shared A frag
    v16bf bg[4], bu[4];
#pragma unroll
    for (int j = 0; j < 4; ++j) {                        // all 8 loads first
      const int ro = (j * 16 + lrow) * BT_STRIDE + khalf * 8;
      const __bf16* pg = &Btg[cur][0] + ro;
      bg[j] = ld_pair(pg, pg + 16);
      const __bf16* pu = &Btu[cur][0] + ro;
      bu[j] = ld_pair(pu, pu + 16);
    }
#pragma unroll
    for (int j = 0; j < 4; ++j) {                        // then 8 WMMAs
      accg[j] = wmma_bf16(a, bg[j], accg[j]);
      accu[j] = wmma_bf16(a, bu[j], accu[j]);
    }
    wait_async0();
    __syncthreads();
    cur ^= 1;
  }

  const int rbase = khalf * 8;
#pragma unroll
  for (int j = 0; j < 4; ++j) {
    const int col = n0 + j * 16 + lrow;
    const float bg = bgate[col];
#pragma unroll
    for (int r = 0; r < 8; ++r) {
      size_t idx = (size_t)(m0 + rbase + r) * DIM_ + col;
      gate[idx] = 1.f / (1.f + expf(-(accg[j][r] + bg)));
      u[idx]    = accu[j][r];
    }
  }
}

// ---------------- recurrence -----------------------------------------------
// 16 persistent WGs (128 threads = 4 waves). WG g owns columns [64g,64g+64),
// keeps A[64g:64g+64,:] in LDS (~129 KB padded) and stages S(t) (32 KB) into
// LDS each step via async copies, so the 16-iter WMMA loop is all-LDS.
// Two accumulators break the serial WMMA D->C dependency chain.
// S is 16x1024 bf16 in global (rows 8..15 stay zero = WMMA pad), ping-ponged;
// one monotonic atomic-counter barrier per step.
__global__ void k_recur(const void* Av, const float* __restrict__ gate,
                        const float* __restrict__ u, const void* inpv,
                        void* Sv, void* statesv,
                        unsigned* __restrict__ cnt) {
  extern __shared__ char smem[];
  __bf16* Asub = (__bf16*)smem;                       // [64][AS_STRIDE]
  __bf16* Slds = (__bf16*)smem + 64 * AS_STRIDE;      // [16][AS_STRIDE]
  const __bf16* __restrict__ A   = (const __bf16*)Av;
  const __bf16* __restrict__ inp = (const __bf16*)inpv;
  __bf16* __restrict__ Sbase  = (__bf16*)Sv;          // 2 x [16][1024]
  __bf16* __restrict__ states = (__bf16*)statesv;
  const int e0   = blockIdx.x * 64;
  const int wave = threadIdx.x >> 5, lane = threadIdx.x & 31;
  const int lrow = lane & 15, khalf = lane >> 4;
  const unsigned asbase = lds_addr(Asub);
  const unsigned slbase = lds_addr(Slds);

  // Stage this WG's A slice into LDS once (async, padded rows).
  for (int i = 0; i < 64; ++i) {
    const int c = threadIdx.x + i * 128;              // 8192 chunks of 8 elems
    const int row = c >> 7, cc = c & 127;
    async_copy16(asbase + (unsigned)((row * AS_STRIDE + cc * 8) * 2),
                 A + (size_t)(e0 + row) * DIM_ + cc * 8);
  }
  wait_async0();
  __syncthreads();

  const int j = wave;                                  // n-tile 0..3
  const __bf16* bsrc = Asub + (size_t)(j * 16 + lrow) * AS_STRIDE + khalf * 8;
  const __bf16* arow = Slds + (size_t)lrow * AS_STRIDE + khalf * 8;
  const int colg = e0 + j * 16 + lrow;

  for (int t = 0; t < SEQ_; ++t) {
    const __bf16* Sr = Sbase + (size_t)(t & 1) * 16 * DIM_;
    __bf16*       Sw = Sbase + (size_t)((t + 1) & 1) * 16 * DIM_;

    // stage S(t) 16x1024 into LDS (2048 async 16B chunks over 128 threads)
#pragma unroll
    for (int i = 0; i < 16; ++i) {
      const int c = threadIdx.x + i * 128;
      const int row = c >> 7, cc = c & 127;
      async_copy16(slbase + (unsigned)((row * AS_STRIDE + cc * 8) * 2),
                   Sr + (size_t)row * DIM_ + cc * 8);
    }
    wait_async0();
    __syncthreads();

    v8f acc0 = vzero8(), acc1 = vzero8();              // break RAW chain
    for (int k0 = 0; k0 < DIM_; k0 += 64) {
      v16bf a0 = ld_pair(arow + k0,      arow + k0 + 16);
      v16bf b0 = ld_pair(bsrc + k0,      bsrc + k0 + 16);
      v16bf a1 = ld_pair(arow + k0 + 32, arow + k0 + 48);
      v16bf b1 = ld_pair(bsrc + k0 + 32, bsrc + k0 + 48);
      acc0 = wmma_bf16(a0, b0, acc0);
      acc1 = wmma_bf16(a1, b1, acc1);
    }
    v8f acc = acc0 + acc1;

    if (lane < 16) {                                   // rows 0..7 = batches
#pragma unroll
      for (int r = 0; r < BATCH_; ++r) {
        size_t idx = ((size_t)r * SEQ_ + t) * DIM_ + colg;
        float pre = acc[r] + u[idx];
        float g   = gate[idx];
        float iv  = (float)inp[idx];
        float ns  = g * tanhf(pre) + (1.f - g) * iv;
        states[idx] = (__bf16)ns;
        Sw[(size_t)r * DIM_ + colg] = (__bf16)ns;
      }
    }

    __syncthreads();                                   // all waves done
    if (threadIdx.x == 0) {
      __threadfence();                                 // publish Sw
      atomicAdd(cnt, 1u);
      const unsigned target = (unsigned)NWG_REC * (unsigned)(t + 1);
      while (__hip_atomic_load(cnt, __ATOMIC_ACQUIRE,
                               __HIP_MEMORY_SCOPE_AGENT) < target)
        __builtin_amdgcn_s_sleep(1);
    }
    __syncthreads();
  }
}

// ---------------------------------------------------------------------------
extern "C" void kernel_launch(void* const* d_in, const int* in_sizes, int n_in,
                              void* d_out, int out_size, void* d_ws, size_t ws_size,
                              hipStream_t stream) {
  (void)in_sizes; (void)n_in; (void)out_size; (void)ws_size;
  const float* x      = (const float*)d_in[0];
  const float* A      = (const float*)d_in[1];
  const float* Bm     = (const float*)d_in[2];
  const float* W_in   = (const float*)d_in[3];
  const float* b_in   = (const float*)d_in[4];
  const float* W_gate = (const float*)d_in[5];
  const float* b_gate = (const float*)d_in[6];
  const float* W_out  = (const float*)d_in[7];
  const float* b_out  = (const float*)d_in[8];

  const size_t TD = (size_t)BATCH_ * SEQ_ * DIM_;      // 33,554,432
  const size_t DD = (size_t)DIM_ * DIM_;               // 1,048,576

  char* w = (char*)d_ws;                               // ~481 MB total
  void*  x_bf   = w; w += TD * 2;
  void*  inp_bf = w; w += TD * 2;
  float* gate   = (float*)w; w += TD * 4;
  float* u      = (float*)w; w += TD * 4;
  void*  st_bf  = w; w += TD * 2;
  void*  A_bf   = w; w += DD * 2;
  void*  B_bf   = w; w += DD * 2;
  void*  Win_bf = w; w += DD * 2;
  void*  Wg_bf  = w; w += DD * 2;
  void*  Wo_bf  = w; w += DD * 2;
  void*  S      = w; w += 2 * 16 * DIM_ * 2;
  unsigned* cnt = (unsigned*)w;

  k_cvt_bf16<<<dim3((unsigned)(TD / 256)), 256, 0, stream>>>(x, x_bf, (int)TD);
  k_cvt_bf16<<<dim3((unsigned)(DD / 256)), 256, 0, stream>>>(A, A_bf, (int)DD);
  k_cvt_bf16<<<dim3((unsigned)(DD / 256)), 256, 0, stream>>>(Bm, B_bf, (int)DD);
  k_cvt_bf16<<<dim3((unsigned)(DD / 256)), 256, 0, stream>>>(W_in, Win_bf, (int)DD);
  k_cvt_bf16<<<dim3((unsigned)(DD / 256)), 256, 0, stream>>>(W_gate, Wg_bf, (int)DD);
  k_cvt_bf16<<<dim3((unsigned)(DD / 256)), 256, 0, stream>>>(W_out, Wo_bf, (int)DD);
  k_init<<<dim3(128), 256, 0, stream>>>(S, cnt);

  dim3 ggrid((unsigned)(BATCH_ * SEQ_ / 128), DIM_ / 64);
  k_gemm<true>  <<<ggrid, 256, 0, stream>>>(x_bf, Win_bf, b_in, inp_bf);
  k_gemm_dual   <<<ggrid, 256, 0, stream>>>(inp_bf, Wg_bf, B_bf, b_gate, gate, u);
  const unsigned rec_lds = (64 + 16) * AS_STRIDE * 2;  // 165,120 B of 320 KB
  k_recur<<<dim3(NWG_REC), 128, rec_lds, stream>>>(A_bf, gate, u, inp_bf,
                                                   S, st_bf, cnt);
  k_gemm<false> <<<ggrid, 256, 0, stream>>>(st_bf, Wo_bf, b_out, (float*)d_out);
}
